// MultiHeadAttention_56375740727843
// MI455X (gfx1250) — compile-verified, matrix operand order checked by reading, not run on hardware
//
#include <hip/hip_runtime.h>
#include <hip/hip_bf16.h>

typedef __attribute__((ext_vector_type(16))) _Float16 v16h;
typedef __attribute__((ext_vector_type(8)))  float    v8f;

#define B_   2
#define S_   2048
#define D_   512
#define H_   8
#define E_   64              // DQ == DV
#define HS_  (H_ * S_)       // 16384
#define EPS_ 1e-5f

// Fused-attention LDS layout (dynamic shared memory)
#define WROW_      (S_ + 16)                       // padded f16 row (bank stagger)
#define WH_BYTES_  (16 * WROW_ * 2)                // 66048: w weights f16 [16][WROW_]
#define RED_OFF_   WH_BYTES_                       // 512:   float [8 waves][16 rows]
#define PART_OFF_  (RED_OFF_ + 8 * 16 * 4)         // 32768: float [8][16][64]
#define SMEM_BYTES_ (PART_OFF_ + 8 * 16 * 64 * 4)  // 99328 total (<320KB/WGP)

// ---------------------------------------------------------------------------
// WMMA helpers.  CDNA5 16-bit A/B 16x32 fragment layout (ISA 7.12.2):
//   lane L: row/col = L%16, K-base = 8*(L/16);
//   v16h elems 0..7  -> K = base+0..7 ; elems 8..15 -> K = base+16..23
// C/D f32 16x16 layout: lane L holds col N=L%16; VGPR r holds row M=r+8*(L/16).
// ---------------------------------------------------------------------------
static __device__ __forceinline__ v8f wmma16(v16h a, v16h b, v8f c) {
  return __builtin_amdgcn_wmma_f32_16x16x32_f16(
      /*neg_a=*/false, a, /*neg_b=*/false, b,
      /*c_mod=*/(short)0, c, /*reuse_a=*/false, /*reuse_b=*/false);
}

static __device__ __forceinline__ v16h frag_ld_f16(const _Float16* base,
                                                   size_t ld, int r0, int k0) {
  const int lane = threadIdx.x & 31;
  const int hi   = lane >> 4;
  const _Float16* p = base + (size_t)(r0 + (lane & 15)) * ld + (size_t)(k0 + hi * 8);
  v16h f;
#pragma unroll
  for (int e = 0; e < 8; ++e) f[e] = p[e];
#pragma unroll
  for (int e = 0; e < 8; ++e) f[e + 8] = p[e + 16];
  return f;
}

static __device__ __forceinline__ v16h frag_ld_f32(const float* __restrict__ base,
                                                   size_t ld, int r0, int k0) {
  const int lane = threadIdx.x & 31;
  const int hi   = lane >> 4;
  const float* p = base + (size_t)(r0 + (lane & 15)) * ld + (size_t)(k0 + hi * 8);
  v16h f;
#pragma unroll
  for (int e = 0; e < 8; ++e) f[e] = (_Float16)p[e];
#pragma unroll
  for (int e = 0; e < 8; ++e) f[e + 8] = (_Float16)p[e + 16];
  return f;
}

// A-fragment of concat(vec): concat[s, h*64+e] = vec[b,h,s,e], vec f16 [H,S,64].
static __device__ __forceinline__ v16h frag_ld_concat(const _Float16* __restrict__ vb,
                                                      int s0, int k0) {
  const int lane = threadIdx.x & 31;
  const int hi   = lane >> 4;
  const int s    = s0 + (lane & 15);
  const int h    = k0 >> 6;
  const int e0   = (k0 & 63) + hi * 8;
  const _Float16* p = vb + ((size_t)h * S_ + s) * E_ + e0;
  v16h f;
#pragma unroll
  for (int e = 0; e < 8; ++e) f[e] = p[e];
#pragma unroll
  for (int e = 0; e < 8; ++e) f[e + 8] = p[e + 16];
  return f;
}

static __device__ __forceinline__ void store_acc_f16(_Float16* __restrict__ dst, size_t ld,
                                                     int m0, int n0, v8f c) {
  const int lane = threadIdx.x & 31;
  const int hi   = lane >> 4;
  const int n    = n0 + (lane & 15);
#pragma unroll
  for (int r = 0; r < 8; ++r)
    dst[(size_t)(m0 + hi * 8 + r) * ld + n] = (_Float16)c[r];
}

static __device__ __forceinline__ void store_acc_f16_T(_Float16* __restrict__ dst, size_t ldT,
                                                       int m0, int n0, v8f c) {
  const int lane = threadIdx.x & 31;
  const int hi   = lane >> 4;
  _Float16* p = dst + (size_t)(n0 + (lane & 15)) * ldT + (size_t)(m0 + hi * 8);
#pragma unroll
  for (int r = 0; r < 8; ++r) p[r] = (_Float16)c[r];
}

// ---------------------------------------------------------------------------
// Prep: f32 [G, Dk, E] -> f16 [G, E, Dk]
// ---------------------------------------------------------------------------
__global__ void __launch_bounds__(256)
k_cvt_transpose(const float* __restrict__ in, _Float16* __restrict__ out,
                int Dk, int E, int total) {
  int idx = blockIdx.x * 256 + threadIdx.x;
  if (idx >= total) return;
  int de = Dk * E;
  int g  = idx / de;
  int r  = idx - g * de;
  int d  = r / E;
  int e  = r - d * E;
  out[(size_t)g * de + (size_t)e * Dk + d] = (_Float16)in[idx];
}

// ---------------------------------------------------------------------------
// QKV projection (Wt f16 [H,E,D]); transposed=1 stores [b,h,e,s] (for V).
// ---------------------------------------------------------------------------
__global__ void __launch_bounds__(256)
k_proj(const float* __restrict__ x, const _Float16* __restrict__ Wt,
       const float* __restrict__ bias, _Float16* __restrict__ out, int transposed) {
  const int bh   = blockIdx.y;
  const int b    = bh / H_;
  const int h    = bh % H_;
  const int wave = threadIdx.x >> 5;
  const int m0   = blockIdx.x * 128 + wave * 16;
  const float*    xb = x  + (size_t)b * S_ * D_;
  const _Float16* wt = Wt + (size_t)h * E_ * D_;

  v8f acc[4] = {};
  for (int kk = 0; kk < D_; kk += 32) {
    v16h a = frag_ld_f32(xb, D_, m0, kk);
#pragma unroll
    for (int j = 0; j < 4; ++j) {
      v16h bf = frag_ld_f16(wt, D_, j * 16, kk);
      acc[j] = wmma16(a, bf, acc[j]);
    }
  }

  const int lane = threadIdx.x & 31;
#pragma unroll
  for (int j = 0; j < 4; ++j) {
    float bv = bias[h * E_ + j * 16 + (lane & 15)];
#pragma unroll
    for (int r = 0; r < 8; ++r) acc[j][r] += bv;
  }

  if (!transposed) {
    _Float16* ob = out + (size_t)(b * H_ + h) * S_ * E_;
#pragma unroll
    for (int j = 0; j < 4; ++j) store_acc_f16(ob, E_, m0, j * 16, acc[j]);
  } else {
    _Float16* ob = out + (size_t)(b * H_ + h) * E_ * S_;
#pragma unroll
    for (int j = 0; j < 4; ++j) store_acc_f16_T(ob, S_, m0, j * 16, acc[j]);
  }
}

// ---------------------------------------------------------------------------
// Fused attention: scores + softmax + w.V for one (b,h,16-row q tile).
//   Phase 1: wave computes 16x256 score chunk in registers (16 v8f, 32 WMMA).
//   Phase 2: softmax: shfl reductions in each 16-lane half (k lives in lanes),
//            cross-wave via tiny LDS array; w -> NT stores to attns output and
//            f16 row-major LDS (the lane transpose the next GEMM needs).
//   Phase 3: w.V partial per wave from LDS A-fragments x global V fragments,
//            partials reduced through LDS -> vec f16 [b,h,s,64].
// Only one pass over the 268MB attns tensor ever touches HBM.
// ---------------------------------------------------------------------------
__global__ void __launch_bounds__(256)
k_attn(const _Float16* __restrict__ Qp, const _Float16* __restrict__ Kp,
       const _Float16* __restrict__ Vpt, const unsigned char* __restrict__ mask,
       float* __restrict__ attn, _Float16* __restrict__ vec) {
  extern __shared__ char smem[];
  _Float16* wh  = (_Float16*)smem;            // [16][WROW_]
  float*    red = (float*)(smem + RED_OFF_);  // [8][16]
  float*    prt = (float*)(smem + PART_OFF_); // [8][16][64]

  const int bh   = blockIdx.y;
  const int b    = bh / H_;
  const int h    = bh % H_;
  const int q0   = blockIdx.x * 16;
  const int wave = threadIdx.x >> 5;
  const int lane = threadIdx.x & 31;
  const int hi   = lane >> 4;
  const int k0   = wave * 256;                // this wave's 256-wide k chunk

  const _Float16* Qb = Qp  + (size_t)(b * H_ + h) * S_ * E_;
  const _Float16* Kb = Kp  + (size_t)(b * H_ + h) * S_ * E_;
  const _Float16* Vb = Vpt + (size_t)(b * H_ + h) * E_ * S_;
  const unsigned char* mb = mask + (size_t)b * S_ * S_;

  // ---- Phase 1: scores (registers) ----
  v16h a0 = frag_ld_f16(Qb, E_, q0, 0);
  v16h a1 = frag_ld_f16(Qb, E_, q0, 32);

  v8f acc[16];
#pragma unroll
  for (int j = 0; j < 16; ++j) {
    v16h b0 = frag_ld_f16(Kb, E_, k0 + j * 16, 0);
    v16h b1 = frag_ld_f16(Kb, E_, k0 + j * 16, 32);
    v8f c = {};
    c = wmma16(a0, b0, c);
    acc[j] = wmma16(a1, b1, c);
  }

  // Prefetch this wave's V slice for phase 3 (overlaps with softmax).
  __builtin_prefetch(Vb + (size_t)lane * S_ + k0, 0, 1);
  __builtin_prefetch(Vb + (size_t)(lane + 32) * S_ + k0, 0, 1);

  // scale + mask + per-lane row maxima
  float pm[8];
#pragma unroll
  for (int r = 0; r < 8; ++r) pm[r] = -__builtin_inff();
#pragma unroll
  for (int j = 0; j < 16; ++j) {
    const int n = k0 + j * 16 + (lane & 15);
#pragma unroll
    for (int r = 0; r < 8; ++r) {
      const int m = q0 + hi * 8 + r;
      float s = acc[j][r] * 0.125f;                 // 1/sqrt(64)
      if (mb[(size_t)m * S_ + n]) s = -__builtin_inff();
      acc[j][r] = s;
      pm[r] = fmaxf(pm[r], s);
    }
  }

  // ---- Phase 2: softmax ----
  // reduce max across the 16 lanes that share this hi half (k dimension)
#pragma unroll
  for (int msk = 1; msk < 16; msk <<= 1)
#pragma unroll
    for (int r = 0; r < 8; ++r) pm[r] = fmaxf(pm[r], __shfl_xor(pm[r], msk, 32));
  if ((lane & 15) == 0) {
#pragma unroll
    for (int r = 0; r < 8; ++r) red[wave * 16 + hi * 8 + r] = pm[r];
  }
  __syncthreads();
  float gm[8];
#pragma unroll
  for (int r = 0; r < 8; ++r) {
    float t = red[hi * 8 + r];
#pragma unroll
    for (int w = 1; w < 8; ++w) t = fmaxf(t, red[w * 16 + hi * 8 + r]);
    gm[r] = t;
  }
  __syncthreads();   // red[] is about to be reused for sums

  float ps[8] = {0.f, 0.f, 0.f, 0.f, 0.f, 0.f, 0.f, 0.f};
#pragma unroll
  for (int j = 0; j < 16; ++j)
#pragma unroll
    for (int r = 0; r < 8; ++r) {
      float e = __expf(acc[j][r] - gm[r]);
      acc[j][r] = e;
      ps[r] += e;
    }
#pragma unroll
  for (int msk = 1; msk < 16; msk <<= 1)
#pragma unroll
    for (int r = 0; r < 8; ++r) ps[r] += __shfl_xor(ps[r], msk, 32);
  if ((lane & 15) == 0) {
#pragma unroll
    for (int r = 0; r < 8; ++r) red[wave * 16 + hi * 8 + r] = ps[r];
  }
  __syncthreads();
  float inv[8];
#pragma unroll
  for (int r = 0; r < 8; ++r) {
    float t = red[hi * 8 + r];
#pragma unroll
    for (int w = 1; w < 8; ++w) t += red[w * 16 + hi * 8 + r];
    inv[r] = 1.0f / t;
  }

  // normalize; stream f32 to attns (non-temporal: write-once data) and stash
  // f16 row-major copy in LDS for the w.V GEMM's A fragments.
  float* arow = attn + ((size_t)b * S_ + q0) * HS_ + (size_t)h * S_;
#pragma unroll
  for (int j = 0; j < 16; ++j) {
    const int n = k0 + j * 16 + (lane & 15);
#pragma unroll
    for (int r = 0; r < 8; ++r) {
      const int m = hi * 8 + r;                     // local q row
      float wv = acc[j][r] * inv[r];
      __builtin_nontemporal_store(wv, &arow[(size_t)m * HS_ + n]);
      wh[m * WROW_ + n] = (_Float16)wv;
    }
  }
  __syncthreads();

  // ---- Phase 3: partial w.V over this wave's k chunk ----
  v8f oc[4] = {};
#pragma unroll
  for (int kk = 0; kk < 256; kk += 32) {
    v16h aw = frag_ld_f16(wh, WROW_, 0, k0 + kk);   // A from LDS (rows = q)
#pragma unroll
    for (int j = 0; j < 4; ++j) {
      v16h bf = frag_ld_f16(Vb, S_, j * 16, k0 + kk);
      oc[j] = wmma16(aw, bf, oc[j]);
    }
  }
#pragma unroll
  for (int j = 0; j < 4; ++j) {
    const int n = j * 16 + (lane & 15);
#pragma unroll
    for (int r = 0; r < 8; ++r)
      prt[(wave * 16 + hi * 8 + r) * 64 + n] = oc[j][r];
  }
  __syncthreads();

  // reduce 8 wave-partials -> vec f16 [b,h,q0+row,n]
  _Float16* ob = vec + ((size_t)(b * H_ + h) * S_ + q0) * E_;
  for (int i = threadIdx.x; i < 16 * 64; i += 256) {
    const int row = i >> 6, n = i & 63;
    float s = 0.f;
#pragma unroll
    for (int w = 0; w < 8; ++w) s += prt[(w * 16 + row) * 64 + n];
    ob[(size_t)row * E_ + n] = (_Float16)s;
  }
}

// ---------------------------------------------------------------------------
// tmp[b,s,n] = concat(vec)[b,s,:] . WO[:,n] + bO[n]   (WOt f16 [N,K])
// ---------------------------------------------------------------------------
__global__ void __launch_bounds__(256)
k_outproj(const _Float16* __restrict__ vec, const _Float16* __restrict__ WOt,
          const float* __restrict__ bO, float* __restrict__ tmp) {
  const int b    = blockIdx.z;
  const int n0   = blockIdx.y * 64;
  const int wave = threadIdx.x >> 5;
  const int m0   = blockIdx.x * 128 + wave * 16;
  const _Float16* vb = vec + (size_t)b * H_ * S_ * E_;

  v8f acc[4] = {};
  for (int kk = 0; kk < D_; kk += 32) {
    v16h a = frag_ld_concat(vb, m0, kk);
#pragma unroll
    for (int j = 0; j < 4; ++j) {
      v16h bf = frag_ld_f16(WOt, D_, n0 + j * 16, kk);
      acc[j] = wmma16(a, bf, acc[j]);
    }
  }
  const int lane = threadIdx.x & 31;
  const int hi   = lane >> 4;
#pragma unroll
  for (int j = 0; j < 4; ++j) {
    const int n = n0 + j * 16 + (lane & 15);
    float bv = bO[n];
#pragma unroll
    for (int r = 0; r < 8; ++r)
      tmp[((size_t)b * S_ + m0 + hi * 8 + r) * D_ + n] = acc[j][r] + bv;
  }
}

// ---------------------------------------------------------------------------
// out = LayerNorm(tmp + q) * gamma + beta.
// ---------------------------------------------------------------------------
__global__ void __launch_bounds__(256)
k_ln(const float* __restrict__ tmp, const float* __restrict__ q,
     const float* __restrict__ gamma, const float* __restrict__ beta,
     float* __restrict__ out) {
  const size_t row = blockIdx.x;  // b*S + s
  const int tid = threadIdx.x;
  const float* tr = tmp + row * D_;
  const float* qr = q + row * D_;
  float x0 = tr[tid] + qr[tid];
  float x1 = tr[tid + 256] + qr[tid + 256];

  __shared__ float red[256];
  red[tid] = x0 + x1;
  __syncthreads();
  for (int s = 128; s > 0; s >>= 1) {
    if (tid < s) red[tid] += red[tid + s];
    __syncthreads();
  }
  const float mu = red[0] * (1.0f / D_);
  __syncthreads();

  const float d0 = x0 - mu, d1 = x1 - mu;
  red[tid] = d0 * d0 + d1 * d1;
  __syncthreads();
  for (int s = 128; s > 0; s >>= 1) {
    if (tid < s) red[tid] += red[tid + s];
    __syncthreads();
  }
  const float rstd = rsqrtf(red[0] * (1.0f / D_) + EPS_);
  out[row * D_ + tid]       = d0 * rstd * gamma[tid] + beta[tid];
  out[row * D_ + tid + 256] = d1 * rstd * gamma[tid + 256] + beta[tid + 256];
}

// ---------------------------------------------------------------------------
// Host launch
// ---------------------------------------------------------------------------
extern "C" void kernel_launch(void* const* d_in, const int* in_sizes, int n_in,
                              void* d_out, int out_size, void* d_ws, size_t ws_size,
                              hipStream_t stream) {
  const float* q    = (const float*)d_in[0];
  const float* k    = (const float*)d_in[1];
  const float* v    = (const float*)d_in[2];
  const unsigned char* mask = (const unsigned char*)d_in[3];
  const float* WQ   = (const float*)d_in[4];
  const float* bQ   = (const float*)d_in[5];
  const float* WK   = (const float*)d_in[6];
  const float* bK   = (const float*)d_in[7];
  const float* WV   = (const float*)d_in[8];
  const float* bV   = (const float*)d_in[9];
  const float* WO   = (const float*)d_in[10];
  const float* bO   = (const float*)d_in[11];
  const float* gamma= (const float*)d_in[12];
  const float* beta = (const float*)d_in[13];

  char* ws = (char*)d_ws;
  _Float16* WQt = (_Float16*)(ws);                                            // 512 KB
  _Float16* WKt = (_Float16*)(ws + (size_t)(512 << 10));                      // 512 KB
  _Float16* WVt = (_Float16*)(ws + (size_t)(1024 << 10));                     // 512 KB
  _Float16* WOt = (_Float16*)(ws + (size_t)(1536 << 10));                     // 512 KB
  _Float16* Qp  = (_Float16*)(ws + (size_t)(2048 << 10));                     // 4 MB
  _Float16* Kp  = (_Float16*)(ws + (size_t)(2048 << 10) + ((size_t)4 << 20)); // 4 MB
  _Float16* Vpt = (_Float16*)(ws + (size_t)(2048 << 10) + ((size_t)8 << 20)); // 4 MB
  _Float16* vec = (_Float16*)(ws + (size_t)(2048 << 10) + ((size_t)12 << 20));// 4 MB
  float*    tmp = (float*)   (ws + (size_t)(2048 << 10) + ((size_t)16 << 20));// 8 MB

  float* outp = (float*)d_out;
  float* attn = outp + (size_t)B_ * S_ * D_;

  // Weight convert + transpose (f32 [.,K,N] -> f16 [.,N,K])
  k_cvt_transpose<<<1024, 256, 0, stream>>>(WQ, WQt, D_, E_, H_ * D_ * E_);
  k_cvt_transpose<<<1024, 256, 0, stream>>>(WK, WKt, D_, E_, H_ * D_ * E_);
  k_cvt_transpose<<<1024, 256, 0, stream>>>(WV, WVt, D_, E_, H_ * D_ * E_);
  k_cvt_transpose<<<1024, 256, 0, stream>>>(WO, WOt, D_, D_, D_ * D_);

  // Q/K/V projections (V stored transposed for the fused w.V GEMM)
  dim3 gp(S_ / 128, B_ * H_);
  k_proj<<<gp, 256, 0, stream>>>(q, WQt, bQ, Qp, 0);
  k_proj<<<gp, 256, 0, stream>>>(k, WKt, bK, Kp, 0);
  k_proj<<<gp, 256, 0, stream>>>(v, WVt, bV, Vpt, 1);

  // Fused scores + softmax + w.V : single HBM pass over the attns tensor.
  dim3 ga(S_ / 16, B_ * H_);
  k_attn<<<ga, 256, SMEM_BYTES_, stream>>>(Qp, Kp, Vpt, mask, attn, vec);

  // Output projection
  dim3 go(S_ / 128, D_ / 64, B_);
  k_outproj<<<go, 256, 0, stream>>>(vec, WOt, bO, tmp);

  // Residual + LayerNorm -> first region of d_out
  k_ln<<<B_ * S_, 256, 0, stream>>>(tmp, q, gamma, beta, outp);
}
